// Block_8744553415185
// MI455X (gfx1250) — compile-verified, matrix operand order checked by reading, not run on hardware
//
#include <hip/hip_runtime.h>
#include <hip/hip_bf16.h>
#include <stdint.h>

typedef __attribute__((ext_vector_type(2))) float v2f;
typedef __attribute__((ext_vector_type(8))) float v8f;
typedef __attribute__((ext_vector_type(8))) int   v8i;

#define Bb 4
#define Nn 1024
#define Cc 1024
#define Hh 16
#define HD 64
#define O3 3072
#define R1 257
#define R2 17

// ---------------------------------------------------------------------------
// CDNA5 async memory->LDS copy (ASYNCcnt path), per CDNA5 ISA
// GV mode: LDS address taken from the VDST VGPR; global address from a VGPR pair plus offset.
// Generic shared pointers carry the LDS byte offset in addr[31:0].
// ---------------------------------------------------------------------------
__device__ inline void async_b128(void* lds_ptr, const void* gptr) {
  uint32_t l = (uint32_t)(uintptr_t)lds_ptr;
  uint64_t g = (uint64_t)(uintptr_t)gptr;
  asm volatile("global_load_async_to_lds_b128 %0, %1, off"
               :: "v"(l), "v"(g) : "memory");
}
#define WAIT_ASYNC(N) asm volatile("s_wait_asynccnt %0" :: "i"(N) : "memory")

// ---------------------------------------------------------------------------
// Fragment loaders (wave32 WMMA VGPR layouts, CDNA5 ISA §7.12.2)
// ---------------------------------------------------------------------------

// IU8 A/B fragment for 16x16x64: tile stored [rowgroup16][64 K-bytes] as dwords.
// V0..V7 byte offsets {0,4,16,20,32,36,48,52} (+8 for lanes 16-31).
__device__ inline v8i load_i8_frag(const uint32_t* tile, int rowbase, int ld_dw) {
  int lane = threadIdx.x & 31;
  int row  = rowbase + (lane & 15);
  int sel  = (lane >> 4) << 1;
  const uint32_t* p = tile + row * ld_dw + sel;
  v8i a;
  a[0] = (int)p[0];  a[1] = (int)p[1];
  a[2] = (int)p[4];  a[3] = (int)p[5];
  a[4] = (int)p[8];  a[5] = (int)p[9];
  a[6] = (int)p[12]; a[7] = (int)p[13];
  return a;
}

// F32 16x16x4 operand, tile stored [row][k]: lanes 0-15 -> {K0,K1}, 16-31 -> {K2,K3}.
__device__ inline v2f load_f32_frag_nk(const float* tile, int rowbase, int k0, int ld) {
  int lane = threadIdx.x & 31;
  int row  = rowbase + (lane & 15);
  int k    = k0 + ((lane >> 4) << 1);
  v2f r; r[0] = tile[row * ld + k]; r[1] = tile[row * ld + k + 1];
  return r;
}

// F32 16x16x4 B operand when tile is stored [k][n] (e.g. V tile [key][d]).
__device__ inline v2f load_f32_frag_kn(const float* tile, int nbase, int k0, int ld) {
  int lane = threadIdx.x & 31;
  int n    = nbase + (lane & 15);
  int k    = k0 + ((lane >> 4) << 1);
  v2f r; r[0] = tile[k * ld + n]; r[1] = tile[(k + 1) * ld + n];
  return r;
}

// ---------------------------------------------------------------------------
// Quantization kernels
// ---------------------------------------------------------------------------
__global__ __launch_bounds__(256) void rowquant_kernel(const float* __restrict__ src,
                                                       uint32_t* __restrict__ q,
                                                       float* __restrict__ scales) {
  __shared__ float red[256];
  int row = blockIdx.x, t = threadIdx.x;
  const float* p = src + (size_t)row * Cc;
  float vals[4], amax = 0.f;
#pragma unroll
  for (int i = 0; i < 4; ++i) { vals[i] = p[t * 4 + i]; amax = fmaxf(amax, fabsf(vals[i])); }
  red[t] = amax; __syncthreads();
  for (int s = 128; s > 0; s >>= 1) { if (t < s) red[t] = fmaxf(red[t], red[t + s]); __syncthreads(); }
  float scale = fmaxf(red[0], 1e-5f) / 127.f;
  if (t == 0) scales[row] = scale;
  float inv = 1.f / scale;
  uint32_t w = 0;
#pragma unroll
  for (int i = 0; i < 4; ++i) {
    float qv = rintf(vals[i] * inv);
    qv = fminf(fmaxf(qv, -128.f), 127.f);
    w |= ((uint32_t)(((int)qv) & 0xff)) << (8 * i);
  }
  q[(size_t)row * (Cc / 4) + t] = w;
}

__global__ __launch_bounds__(1024) void bquant_kernel(const float* __restrict__ b,
                                                      float* __restrict__ qb) {
  __shared__ float red[1024];
  int t = threadIdx.x;
  float amax = 0.f;
  for (int i = t; i < O3; i += 1024) amax = fmaxf(amax, fabsf(b[i]));
  red[t] = amax; __syncthreads();
  for (int s = 512; s > 0; s >>= 1) { if (t < s) red[t] = fmaxf(red[t], red[t + s]); __syncthreads(); }
  float scale = fmaxf(red[0], 1e-5f) / 127.f;
  for (int i = t; i < O3; i += 1024) {
    float qv = rintf(b[i] / scale);
    qv = fminf(fmaxf(qv, -128.f), 127.f);
    qb[i] = qv * scale;
  }
}

__global__ __launch_bounds__(256) void rowabsmax_kernel(const float* __restrict__ src,
                                                        float* __restrict__ scales) {
  __shared__ float red[256];
  int row = blockIdx.x, t = threadIdx.x;
  const float* p = src + (size_t)row * O3;
  float amax = 0.f;
  for (int i = t; i < O3; i += 256) amax = fmaxf(amax, fabsf(p[i]));
  red[t] = amax; __syncthreads();
  for (int s = 128; s > 0; s >>= 1) { if (t < s) red[t] = fmaxf(red[t], red[t + s]); __syncthreads(); }
  if (t == 0) scales[row] = fmaxf(red[0], 1e-5f) / 127.f;
}

// Pack q_lin's Q and K sections (cols 0..2047) to int8 once: q8[row][512 dwords].
__global__ __launch_bounds__(256) void qlin_pack_kernel(const float* __restrict__ qlin,
                                                        const float* __restrict__ qls,
                                                        uint32_t* __restrict__ q8) {
  int row = blockIdx.x, t = threadIdx.x;
  float inv = 1.f / qls[row];
  const float* p = qlin + (size_t)row * O3;
  for (int dw = t; dw < 512; dw += 256) {
    uint32_t w = 0;
#pragma unroll
    for (int j = 0; j < 4; ++j) {
      float qv = rintf(p[dw * 4 + j] * inv);
      qv = fminf(fmaxf(qv, -128.f), 127.f);
      w |= ((uint32_t)(((int)qv) & 0xff)) << (8 * j);
    }
    q8[(size_t)row * 512 + dw] = w;
  }
}

// ---------------------------------------------------------------------------
// Quantized QKV GEMM (IU8 WMMA), async double-buffered tiles.
// ---------------------------------------------------------------------------
__global__ __launch_bounds__(128) void qkv_i8_gemm_kernel(const uint32_t* __restrict__ qx,
                                                          const float* __restrict__ sx,
                                                          const uint32_t* __restrict__ qw,
                                                          const float* __restrict__ sw,
                                                          const float* __restrict__ qb,
                                                          float* __restrict__ out) {
  __shared__ uint32_t As[2][64 * 16];
  __shared__ uint32_t Bs[2][64 * 16];
  int tid = threadIdx.x, wave = tid >> 5, lane = tid & 31;
  int mb = blockIdx.y * 64, nb = blockIdx.x * 64;
  v8i acc[4] = {};

  auto issue = [&](int kc, int buf) {
    for (int i = tid; i < 256; i += 128) {
      int r = i >> 2, c4 = (i & 3) << 2;
      async_b128(&As[buf][r * 16 + c4], &qx[(size_t)(mb + r) * 256 + kc * 16 + c4]);
      async_b128(&Bs[buf][r * 16 + c4], &qw[(size_t)(nb + r) * 256 + kc * 16 + c4]);
    }
  };

  issue(0, 0);
  for (int kc = 0; kc < 16; ++kc) {
    int buf = kc & 1;
    WAIT_ASYNC(0);          // stage kc resident (each wave waits its own issues)
    __syncthreads();        // all waves: stage kc visible, prev compute done
    if (kc < 15) issue(kc + 1, buf ^ 1);   // overlaps with compute below
    v8i a = load_i8_frag(As[buf], wave * 16, 16);
#pragma unroll
    for (int nt = 0; nt < 4; ++nt) {
      v8i b = load_i8_frag(Bs[buf], nt * 16, 16);
      acc[nt] = __builtin_amdgcn_wmma_i32_16x16x64_iu8(true, a, true, b, acc[nt], false, false);
    }
  }
#pragma unroll
  for (int nt = 0; nt < 4; ++nt)
#pragma unroll
    for (int r = 0; r < 8; ++r) {
      int m = mb + wave * 16 + r + ((lane >> 4) << 3);
      int n = nb + nt * 16 + (lane & 15);
      out[(size_t)m * O3 + n] = (float)acc[nt][r] * sx[m] * sw[n] + qb[n];
    }
}

// ---------------------------------------------------------------------------
// q_attn from pre-packed int8: async tile fills, one IU8 WMMA (K=64) per tile.
// ---------------------------------------------------------------------------
__global__ __launch_bounds__(128) void qattn_kernel(const uint32_t* __restrict__ q8,
                                                    const float* __restrict__ qls,
                                                    float* __restrict__ qattn) {
  __shared__ uint32_t Qs[64 * 16];
  __shared__ uint32_t Ks[64 * 16];
  int tid = threadIdx.x, wave = tid >> 5, lane = tid & 31;
  int bh = blockIdx.z, b = bh >> 4, h = bh & 15;
  int nbase = blockIdx.y * 64;   // query rows
  int mbase = blockIdx.x * 64;   // key rows
  for (int i = tid; i < 256; i += 128) {
    int r = i >> 2, c4 = (i & 3) << 2;
    async_b128(&Qs[r * 16 + c4], &q8[(size_t)(b * Nn + nbase + r) * 512 + h * 16 + c4]);
    async_b128(&Ks[r * 16 + c4], &q8[(size_t)(b * Nn + mbase + r) * 512 + 256 + h * 16 + c4]);
  }
  WAIT_ASYNC(0);
  __syncthreads();
  v8i a = load_i8_frag(Qs, wave * 16, 16);
  v8i acc[4] = {};
#pragma unroll
  for (int nt = 0; nt < 4; ++nt) {
    v8i bfr = load_i8_frag(Ks, nt * 16, 16);
    acc[nt] = __builtin_amdgcn_wmma_i32_16x16x64_iu8(true, a, true, bfr, acc[nt], false, false);
  }
#pragma unroll
  for (int nt = 0; nt < 4; ++nt)
#pragma unroll
    for (int r = 0; r < 8; ++r) {
      int n = nbase + wave * 16 + r + ((lane >> 4) << 3);
      int m = mbase + nt * 16 + (lane & 15);
      float v = (float)acc[nt][r] * qls[b * Nn + n] * qls[b * Nn + m] * 0.125f;
      qattn[((size_t)bh * Nn + n) * Nn + m] = v;
    }
}

// ---------------------------------------------------------------------------
// Exact f32 GEMM via V_WMMA_F32_16X16X4_F32, async double-buffered K-chunks of 32.
// C[M][Ncols] = X[M][1024] @ Wt[Ncols][1024]^T + bias
// ---------------------------------------------------------------------------
__global__ __launch_bounds__(128) void gemm_f32_kernel(const float* __restrict__ X,
                                                       const float* __restrict__ Wt,
                                                       const float* __restrict__ bias,
                                                       float* __restrict__ Cmat, int Ncols) {
  __shared__ float As[2][64 * 32];
  __shared__ float Bs[2][64 * 32];
  int tid = threadIdx.x, wave = tid >> 5, lane = tid & 31;
  int mb = blockIdx.y * 64, nb = blockIdx.x * 64;
  v8f acc[4] = {};

  auto issue = [&](int kc, int buf) {
    for (int i = tid; i < 512; i += 128) {
      int r = i >> 3, c4 = (i & 7) << 2;
      async_b128(&As[buf][r * 32 + c4], &X[(size_t)(mb + r) * 1024 + kc * 32 + c4]);
      async_b128(&Bs[buf][r * 32 + c4], &Wt[(size_t)(nb + r) * 1024 + kc * 32 + c4]);
    }
  };

  issue(0, 0);
  for (int kc = 0; kc < 32; ++kc) {
    int buf = kc & 1;
    WAIT_ASYNC(0);
    __syncthreads();
    if (kc < 31) issue(kc + 1, buf ^ 1);   // copy next stage during WMMA below
#pragma unroll
    for (int k4 = 0; k4 < 8; ++k4) {
      v2f a = load_f32_frag_nk(As[buf], wave * 16, k4 * 4, 32);
#pragma unroll
      for (int nt = 0; nt < 4; ++nt) {
        v2f b = load_f32_frag_nk(Bs[buf], nt * 16, k4 * 4, 32);
        acc[nt] = __builtin_amdgcn_wmma_f32_16x16x4_f32(false, a, false, b,
                                                        (short)0, acc[nt], false, false);
      }
    }
  }
#pragma unroll
  for (int nt = 0; nt < 4; ++nt)
#pragma unroll
    for (int r = 0; r < 8; ++r) {
      int m = mb + wave * 16 + r + ((lane >> 4) << 3);
      int n = nb + nt * 16 + (lane & 15);
      Cmat[(size_t)m * Ncols + n] = acc[nt][r] + bias[n];
    }
}

// ---------------------------------------------------------------------------
// Flash attention (exact path): async K/V tile fills, online softmax,
// S and P.V via f32 WMMA; P reuses the K tile's LDS after S is consumed.
// ---------------------------------------------------------------------------
__global__ __launch_bounds__(128) void attn_kernel(const float* __restrict__ lin,
                                                   float* __restrict__ Obuf) {
  __shared__ float Qs[64 * 64];
  __shared__ float Ks[64 * 64];   // reused as P buffer
  __shared__ float Vs[64 * 64];
  int tid = threadIdx.x, wave = tid >> 5, lane = tid & 31;
  int bh = blockIdx.y, b = bh >> 4, h = bh & 15;
  int qbase = blockIdx.x * 64;

  // async fill of the Q tile (waited together with the first K/V stage)
  for (int i = tid; i < 1024; i += 128) {
    int r = i >> 4, c4 = (i & 15) << 2;
    async_b128(&Qs[r * 64 + c4], &lin[(size_t)(b * Nn + qbase + r) * O3 + h * HD + c4]);
  }

  float mrow[8], lrow[8];
#pragma unroll
  for (int r = 0; r < 8; ++r) { mrow[r] = -1e30f; lrow[r] = 0.f; }
  v8f oacc[4] = {};

  for (int kt = 0; kt < 16; ++kt) {
    __syncthreads();   // all waves finished reading Ks/Vs (and P) of prev tile
    for (int i = tid; i < 1024; i += 128) {
      int r = i >> 4, c4 = (i & 15) << 2;
      int krow = b * Nn + kt * 64 + r;
      async_b128(&Ks[r * 64 + c4], &lin[(size_t)krow * O3 + Cc + h * HD + c4]);
      async_b128(&Vs[r * 64 + c4], &lin[(size_t)krow * O3 + 2 * Cc + h * HD + c4]);
    }
    WAIT_ASYNC(0);
    __syncthreads();

    // S strip: 16 rows x 64 keys, K = hd = 64
    v8f s[4] = {};
#pragma unroll
    for (int k4 = 0; k4 < 16; ++k4) {
      v2f a = load_f32_frag_nk(Qs, wave * 16, k4 * 4, 64);
#pragma unroll
      for (int nt = 0; nt < 4; ++nt) {
        v2f bfr = load_f32_frag_nk(Ks, nt * 16, k4 * 4, 64);
        s[nt] = __builtin_amdgcn_wmma_f32_16x16x4_f32(false, a, false, bfr,
                                                      (short)0, s[nt], false, false);
      }
    }
    __syncthreads();   // Ks fully consumed; safe to overwrite with P

    float* Pw = Ks + wave * 16 * 64;
#pragma unroll
    for (int r = 0; r < 8; ++r) {
      float tm = -1e30f;
#pragma unroll
      for (int nt = 0; nt < 4; ++nt) tm = fmaxf(tm, s[nt][r] * 0.125f);
      for (int off = 1; off < 16; off <<= 1) tm = fmaxf(tm, __shfl_xor(tm, off, 32));
      float mnew  = fmaxf(mrow[r], tm);
      float alpha = __expf(mrow[r] - mnew);
      float psum = 0.f;
      int wrow = r + ((lane >> 4) << 3);
#pragma unroll
      for (int nt = 0; nt < 4; ++nt) {
        float p = __expf(s[nt][r] * 0.125f - mnew);
        psum += p;
        Pw[wrow * 64 + nt * 16 + (lane & 15)] = p;
      }
      for (int off = 1; off < 16; off <<= 1) psum += __shfl_xor(psum, off, 32);
      lrow[r] = lrow[r] * alpha + psum;
      mrow[r] = mnew;
#pragma unroll
      for (int nt = 0; nt < 4; ++nt) oacc[nt][r] *= alpha;
    }
    __syncthreads();

    // O += P @ V
#pragma unroll
    for (int k4 = 0; k4 < 16; ++k4) {
      v2f a = load_f32_frag_nk(Pw, 0, k4 * 4, 64);
#pragma unroll
      for (int nt = 0; nt < 4; ++nt) {
        v2f bfr = load_f32_frag_kn(Vs, nt * 16, k4 * 4, 64);
        oacc[nt] = __builtin_amdgcn_wmma_f32_16x16x4_f32(false, a, false, bfr,
                                                         (short)0, oacc[nt], false, false);
      }
    }
  }

#pragma unroll
  for (int nt = 0; nt < 4; ++nt)
#pragma unroll
    for (int r = 0; r < 8; ++r) {
      int row = qbase + wave * 16 + r + ((lane >> 4) << 3);
      int d   = nt * 16 + (lane & 15);
      Obuf[(size_t)(b * Nn + row) * Cc + h * HD + d] = oacc[nt][r] / lrow[r];
    }
}

// ---------------------------------------------------------------------------
// Topk path (tiny: 17 sequence rows)
// ---------------------------------------------------------------------------
__global__ __launch_bounds__(256) void topk_lin_kernel(const float* __restrict__ x,
                                                       const float* __restrict__ W,
                                                       const float* __restrict__ bq,
                                                       float* __restrict__ tlin) {
  int bn = blockIdx.x, b = bn / R2, n = bn % R2;
  const float* xr = x + ((size_t)b * Nn + n) * Cc;
  for (int o = threadIdx.x; o < 2048; o += 256) {
    const float* wr = W + (size_t)o * Cc;
    float acc = 0.f;
    for (int c = 0; c < R1; ++c) acc += xr[c] * wr[c];
    tlin[((size_t)b * R2 + n) * 2048 + o] = acc + bq[o];
  }
}

__global__ __launch_bounds__(256) void topk_attn_kernel(const float* __restrict__ tlin,
                                                        float* __restrict__ tattn) {
  int idx = blockIdx.x * blockDim.x + threadIdx.x;
  if (idx >= Bb * Hh * R2 * R2) return;
  int m = idx % R2, n = (idx / R2) % R2, h = (idx / (R2 * R2)) % Hh, b = idx / (R2 * R2 * Hh);
  const float* qr = tlin + ((size_t)b * R2 + n) * 2048 + h * HD;
  const float* kr = tlin + ((size_t)b * R2 + m) * 2048 + Cc + h * HD;
  float acc = 0.f;
#pragma unroll
  for (int d = 0; d < HD; ++d) acc += qr[d] * kr[d];
  tattn[idx] = acc * 0.125f;
}

// ---------------------------------------------------------------------------
extern "C" void kernel_launch(void* const* d_in, const int* in_sizes, int n_in,
                              void* d_out, int out_size, void* d_ws, size_t ws_size,
                              hipStream_t stream) {
  (void)in_sizes; (void)n_in; (void)out_size; (void)ws_size;
  const float* x     = (const float*)d_in[0];
  const float* Wqkv  = (const float*)d_in[1];
  const float* bqkv  = (const float*)d_in[2];
  const float* Wproj = (const float*)d_in[3];
  const float* bproj = (const float*)d_in[4];

  float* out   = (float*)d_out;                                  // [4,1024,1024]
  float* qattn = out + (size_t)Bb * Nn * Cc;                     // [4,16,1024,1024]
  float* tattn = qattn + (size_t)Bb * Hh * Nn * Nn;              // [4,16,17,17]

  char* ws = (char*)d_ws;
  size_t off = 0;
  auto carve = [&](size_t bytes) -> void* {
    void* p = ws + off;
    off = (off + bytes + 255) & ~(size_t)255;
    return p;
  };
  uint32_t* qx   = (uint32_t*)carve((size_t)4096 * 1024);            // int8 x
  float*    sx   = (float*)carve(4096 * sizeof(float));
  uint32_t* qw   = (uint32_t*)carve((size_t)3072 * 1024);            // int8 W_qkv
  float*    sw   = (float*)carve(3072 * sizeof(float));
  float*    qb   = (float*)carve(3072 * sizeof(float));
  float*    qlin = (float*)carve((size_t)4096 * 3072 * sizeof(float));
  float*    qls  = (float*)carve(4096 * sizeof(float));
  uint32_t* q8   = (uint32_t*)carve((size_t)4096 * 2048);            // int8 q_lin (Q,K)
  float*    lin  = (float*)carve((size_t)4096 * 3072 * sizeof(float));
  float*    Obuf = (float*)carve((size_t)4096 * 1024 * sizeof(float));
  float*    tlin = (float*)carve((size_t)Bb * R2 * 2048 * sizeof(float));

  // 1. quantization
  rowquant_kernel<<<4096, 256, 0, stream>>>(x, qx, sx);
  rowquant_kernel<<<3072, 256, 0, stream>>>(Wqkv, qw, sw);
  bquant_kernel<<<1, 1024, 0, stream>>>(bqkv, qb);

  // 2. quantized QKV GEMM (IU8 WMMA) + per-row scale + int8 pack of q_lin
  qkv_i8_gemm_kernel<<<dim3(48, 64), 128, 0, stream>>>(qx, sx, qw, sw, qb, qlin);
  rowabsmax_kernel<<<4096, 256, 0, stream>>>(qlin, qls);
  qlin_pack_kernel<<<4096, 256, 0, stream>>>(qlin, qls, q8);

  // 3. q_attn (IU8 WMMA, K = hd = 64 in a single instruction)
  qattn_kernel<<<dim3(16, 16, 64), 128, 0, stream>>>(q8, qls, qattn);

  // 4. topk path (tiny)
  topk_lin_kernel<<<Bb * R2, 256, 0, stream>>>(x, Wqkv, bqkv, tlin);
  topk_attn_kernel<<<(Bb * Hh * R2 * R2 + 255) / 256, 256, 0, stream>>>(tlin, tattn);

  // 5. exact path: lin GEMM -> flash attention -> proj GEMM (f32 WMMA)
  gemm_f32_kernel<<<dim3(48, 64), 128, 0, stream>>>(x, Wqkv, bqkv, lin, O3);
  attn_kernel<<<dim3(16, 64), 128, 0, stream>>>(lin, Obuf);
  gemm_f32_kernel<<<dim3(16, 64), 128, 0, stream>>>(Obuf, Wproj, bproj, out, Cc);
}
